// CNNSelfAttention_37503654429225
// MI455X (gfx1250) — matrix operand order repaired and checked by reading, not compiled, not run on hardware
//
#include <hip/hip_runtime.h>
#include <hip/hip_bf16.h>

// ---------------------------------------------------------------------------
// CNN tile self-attention for MI455X (gfx1250): all GEMMs on
// v_wmma_f32_16x16x32_bf16, LDS-resident im2col slabs in B-fragment layout,
// async global->LDS staging for the final conv. conv_qkv K-loop is outermost
// with explicit 12-tile register blocking so B-fragments never spill.
// ---------------------------------------------------------------------------

typedef __bf16 bf16_t;
typedef bf16_t v8bf  __attribute__((ext_vector_type(8)));
typedef bf16_t v16bf __attribute__((ext_vector_type(16)));
typedef float  v8f   __attribute__((ext_vector_type(8)));

#define B_    4
#define NH_   8
#define T_    64
#define C_    64
#define HW_   256
#define CHW_  16384   // C_*HW_
#define KQ_   576     // C_*9  (im2col depth, qkv conv)
#define KU_   4608    // 512*9 (im2col depth, output conv)
#define SLABP 584     // padded slab row stride (elements); 584*2 = 73*16 bytes

__device__ __forceinline__ unsigned short f2bf(float f) {
  unsigned u = __builtin_bit_cast(unsigned, f);
  u += 0x7FFFu + ((u >> 16) & 1u);           // round-to-nearest-even
  return (unsigned short)(u >> 16);
}
__device__ __forceinline__ bf16_t us2bf(unsigned short u) {
  return __builtin_bit_cast(bf16_t, u);
}
__device__ __forceinline__ v16bf cat16(v8bf lo, v8bf hi) {
  v16bf r;
  #pragma unroll
  for (int j = 0; j < 8; ++j) { r[j] = lo[j]; r[8 + j] = hi[j]; }
  return r;
}

// ---------------------------------------------------------------------------
// Weight packing: Wq|Wk|Wv -> bf16 [1536][576], K index = c*9 + tap.
// 1/scale (scale = (H*W)^0.25 = 4) folded into q and k weight rows.
// ---------------------------------------------------------------------------
__global__ void pack_wcat(const float* __restrict__ Wq,
                          const float* __restrict__ Wk,
                          const float* __restrict__ Wv,
                          unsigned short* __restrict__ wcat) {
  int idx = blockIdx.x * blockDim.x + threadIdx.x;
  if (idx >= 1536 * KQ_) return;
  int m   = idx / KQ_;
  int kk  = idx - m * KQ_;
  int c   = kk / 9, tap = kk - c * 9;
  int o   = m & 511;
  const float* src = (m < 512) ? Wq : (m < 1024 ? Wk : Wv);
  float v = src[(o * C_ + c) * 9 + tap];
  if (m < 1024) v *= 0.25f;
  wcat[idx] = f2bf(v);
}

__global__ void pack_wu(const float* __restrict__ Wu,
                        unsigned short* __restrict__ wub) {
  int idx = blockIdx.x * blockDim.x + threadIdx.x;
  if (idx >= 64 * KU_) return;
  int m  = idx / KU_;
  int kk = idx - m * KU_;
  int c  = kk / 9, tap = kk - c * 9;
  wub[idx] = f2bf(Wu[(m * 512 + c) * 9 + tap]);
}

// ---------------------------------------------------------------------------
// Build one 16-column im2col slab (bf16, [16][SLABP], k contiguous) in LDS
// from a 64ch x 16x16 bf16 image already resident in LDS. Branch-free:
// clamped LDS read + select-zero.
// ---------------------------------------------------------------------------
__device__ __forceinline__ void build_slab(const unsigned short* __restrict__ ldsImg,
                                           unsigned short* __restrict__ slab,
                                           int n0, int tid, int nworkers) {
  const int col = tid & 15;
  const int n   = n0 + col;
  const int ny  = n >> 4, nx = n & 15;
  for (int kk = tid >> 4; kk < KQ_; kk += nworkers) {
    int c   = kk / 9, tap = kk - c * 9;
    int ky  = tap / 3, kx = tap - ky * 3;
    int y   = ny + ky - 1, xx = nx + kx - 1;
    bool inb = ((unsigned)y < 16u) & ((unsigned)xx < 16u);
    unsigned short u = ldsImg[c * HW_ + (y & 15) * 16 + (xx & 15)];
    slab[col * SLABP + kk] = inb ? u : (unsigned short)0;
  }
}

// ---------------------------------------------------------------------------
// Fused q/k/v conv as implicit GEMM. Block = (image, N-tile). Input image
// staged bf16 in LDS, im2col slab built once. K-loop outer; each wave holds
// 12 accumulator tiles (M-tiles wv, wv+8, ..., wv+88) so each B-fragment is
// loaded once from LDS and consumed by 12 WMMAs. Outputs:
//   q,k : [b][head][t][chw]    (row-major over K for the score GEMM)
//   vT  : [b][head][chw][s]    (column-contiguous B-operand for P*V)
// ---------------------------------------------------------------------------
__global__ void conv_qkv(const float* __restrict__ x,
                         const unsigned short* __restrict__ wcat,
                         unsigned short* __restrict__ q,
                         unsigned short* __restrict__ k,
                         unsigned short* __restrict__ vt) {
  __shared__ unsigned short ldsX[C_ * HW_];        // 32 KB
  __shared__ unsigned short slab[16 * SLABP];      // 18.25 KB
  const int img = blockIdx.x;                      // 0..255
  const int n0  = blockIdx.y * 16;                 // N-tile base
  const int tid = threadIdx.x;

  const float* xs = x + (size_t)img * CHW_;
  for (int i = tid; i < CHW_; i += 256) ldsX[i] = f2bf(xs[i]);
  __syncthreads();
  build_slab(ldsX, slab, n0, tid, 16);
  __syncthreads();

  const int lane = tid & 31;
  const int wv   = tid >> 5;                       // 8 waves
  const int ln15 = lane & 15;
  const int lhi  = lane >> 4;                      // 0/1
  const int b = img >> 6, t = img & 63;
  const int n = n0 + ln15;
  const unsigned short* bptr = slab + ln15 * SLABP + lhi * 16;
  // A base for M-tile (wv + 8*i): row stride per i = 8*16*KQ_ elements
  const unsigned short* abase = wcat + (size_t)(wv * 16 + ln15) * KQ_ + lhi * 16;

  v8f acc[12];
  #pragma unroll
  for (int i = 0; i < 12; ++i)
    acc[i] = (v8f){0.f, 0.f, 0.f, 0.f, 0.f, 0.f, 0.f, 0.f};

  #pragma unroll 1
  for (int ks = 0; ks < KQ_ / 32; ++ks) {
    const int kb = ks * 32;
    const v16bf Bm = cat16(*(const v8bf*)(bptr + kb),
                           *(const v8bf*)(bptr + kb + 8));
    #pragma unroll
    for (int i = 0; i < 12; ++i) {
      const unsigned short* ap = abase + (size_t)i * (8 * 16 * KQ_) + kb;
      v16bf A = cat16(*(const v8bf*)(ap), *(const v8bf*)(ap + 8));
      acc[i] = __builtin_amdgcn_wmma_f32_16x16x32_bf16(false, A, false, Bm,
                                                       (short)0, acc[i],
                                                       false, false);
    }
  }

  // Scatter D: lane L -> (M = r + 8*(L>=16), N = L&15)
  #pragma unroll 1
  for (int i = 0; i < 12; ++i) {
    const int m0 = (wv + 8 * i) * 16;
    #pragma unroll
    for (int r = 0; r < 8; ++r) {
      int m    = m0 + r + lhi * 8;
      int t3   = m >> 9;                           // 0=q 1=k 2=v
      int o    = m & 511;
      int head = o >> 6, c = o & 63;
      int bh   = b * NH_ + head;
      unsigned short val = f2bf(acc[i][r]);
      if (t3 == 0)
        q[(size_t)(bh * T_ + t) * CHW_ + c * HW_ + n] = val;
      else if (t3 == 1)
        k[(size_t)(bh * T_ + t) * CHW_ + c * HW_ + n] = val;
      else
        vt[((size_t)bh * CHW_ + c * HW_ + n) * T_ + t] = val;
    }
  }
}

// ---------------------------------------------------------------------------
// S = Q*K^T (64x64 per (b,head), K=16384) then row softmax -> P (bf16).
// 16 waves, one 16x16 S-tile each, 512 WMMAs deep, contiguous b128 loads.
// ---------------------------------------------------------------------------
__global__ void scores_softmax(const unsigned short* __restrict__ q,
                               const unsigned short* __restrict__ k,
                               unsigned short* __restrict__ p) {
  __shared__ float sl[T_ * T_];                    // 16 KB
  const int bh   = blockIdx.x;
  const int tid  = threadIdx.x;
  const int lane = tid & 31, wv = tid >> 5;        // 16 waves
  const int tm   = wv >> 2, tn = wv & 3;
  const int ln15 = lane & 15, lhi = lane >> 4;

  const unsigned short* qrow =
      q + (size_t)(bh * T_ + tm * 16 + ln15) * CHW_ + lhi * 16;
  const unsigned short* krow =
      k + (size_t)(bh * T_ + tn * 16 + ln15) * CHW_ + lhi * 16;

  v8f acc = {0.f, 0.f, 0.f, 0.f, 0.f, 0.f, 0.f, 0.f};
  #pragma unroll 2
  for (int ks = 0; ks < CHW_ / 32; ++ks) {
    const int kb = ks * 32;
    __builtin_prefetch(qrow + kb + 512, 0, 0);
    __builtin_prefetch(krow + kb + 512, 0, 0);
    v16bf A  = cat16(*(const v8bf*)(qrow + kb), *(const v8bf*)(qrow + kb + 8));
    v16bf Bm = cat16(*(const v8bf*)(krow + kb), *(const v8bf*)(krow + kb + 8));
    acc = __builtin_amdgcn_wmma_f32_16x16x32_bf16(false, A, false, Bm,
                                                  (short)0, acc, false, false);
  }
  #pragma unroll
  for (int r = 0; r < 8; ++r)
    sl[(tm * 16 + r + lhi * 8) * T_ + tn * 16 + ln15] = acc[r];
  __syncthreads();

  if (tid < T_) {
    const float* row = sl + tid * T_;
    float mx = -1e30f;
    for (int s = 0; s < T_; ++s) mx = fmaxf(mx, row[s]);
    float sum = 0.f;
    for (int s = 0; s < T_; ++s) sum += __expf(row[s] - mx);
    float inv = 1.f / sum;
    unsigned short* prow = p + (size_t)(bh * T_ + tid) * T_;
    for (int s = 0; s < T_; ++s) prow[s] = f2bf(__expf(row[s] - mx) * inv);
  }
}

// ---------------------------------------------------------------------------
// O = P*V  (M=64 t's, N=16384 features, K=64), output into per-image
// (512ch x 256) bf16 layout for the final conv.
// ---------------------------------------------------------------------------
__global__ void attn_out(const unsigned short* __restrict__ p,
                         const unsigned short* __restrict__ vt,
                         unsigned short* __restrict__ O) {
  const int ng   = blockIdx.x;                     // 0..511
  const int bh   = blockIdx.y;                     // 0..31
  const int tid  = threadIdx.x;
  const int lane = tid & 31, wv = tid >> 5;        // 8 waves
  const int tm   = wv & 3;
  const int nt   = ng * 2 + (wv >> 2);             // N-tile 0..1023
  const int ln15 = lane & 15, lhi = lane >> 4;

  const unsigned short* arow =
      p + (size_t)(bh * T_ + tm * 16 + ln15) * T_ + lhi * 16;
  const unsigned short* bcol =
      vt + ((size_t)bh * CHW_ + nt * 16 + ln15) * T_ + lhi * 16;

  v8f acc = {0.f, 0.f, 0.f, 0.f, 0.f, 0.f, 0.f, 0.f};
  #pragma unroll
  for (int ks = 0; ks < 2; ++ks) {
    const int kb = ks * 32;
    v16bf A  = cat16(*(const v8bf*)(arow + kb), *(const v8bf*)(arow + kb + 8));
    v16bf Bm = cat16(*(const v8bf*)(bcol + kb), *(const v8bf*)(bcol + kb + 8));
    acc = __builtin_amdgcn_wmma_f32_16x16x32_bf16(false, A, false, Bm,
                                                  (short)0, acc, false, false);
  }
  const int b = bh >> 3, head = bh & 7;
  const int f = nt * 16 + ln15, c = f >> 8, sp = f & 255;
  #pragma unroll
  for (int r = 0; r < 8; ++r) {
    int t   = tm * 16 + r + lhi * 8;
    int img = b * T_ + t;
    O[((size_t)img * 512 + head * 64 + c) * HW_ + sp] = f2bf(acc[r]);
  }
}

// ---------------------------------------------------------------------------
// Final conv 512->64 as implicit GEMM. Block = (image, N-tile), 4 waves,
// one 16-row M-tile each. 8 phases of 64 input channels: each phase stages
// the bf16 activation chunk into LDS with global_load_async_to_lds_b128
// (ASYNCcnt), builds the im2col slab, runs 18 WMMAs per wave.
// ---------------------------------------------------------------------------
__global__ void conv_out(const unsigned short* __restrict__ O,
                         const unsigned short* __restrict__ wub,
                         float* __restrict__ out) {
  __shared__ unsigned short ldsO[64 * HW_];        // 32 KB
  __shared__ unsigned short slab[16 * SLABP];      // 18.25 KB
  const int img  = blockIdx.x;                     // 0..255
  const int n0   = blockIdx.y * 16;                // N-tile base
  const int tid  = threadIdx.x;                    // 128 threads, 4 waves
  const int lane = tid & 31, wv = tid >> 5;
  const int ln15 = lane & 15, lhi = lane >> 4;
  const int n    = n0 + ln15;

  const unsigned ldsBase = (unsigned)(size_t)(&ldsO[0]);
  v8f acc = {0.f, 0.f, 0.f, 0.f, 0.f, 0.f, 0.f, 0.f};

  const unsigned short* bptr = slab + ln15 * SLABP + lhi * 16;
  const unsigned short* aRowBase = wub + (size_t)(wv * 16 + ln15) * KU_ + lhi * 16;

  #pragma unroll 1
  for (int ph = 0; ph < 8; ++ph) {
    __syncthreads();
    // Async-stage 64 channels (32 KB, fully contiguous) into LDS.
    {
      const unsigned short* src = O + ((size_t)img * 512 + ph * 64) * HW_;
      for (int i = tid; i < 2048; i += 128) {      // 2048 x 16B transfers
        unsigned long long ga = (unsigned long long)(size_t)(src) + (unsigned long long)i * 16u;
        unsigned           la = ldsBase + (unsigned)i * 16u;
        asm volatile("global_load_async_to_lds_b128 %0, %1, off"
                     :: "v"(la), "v"(ga) : "memory");
      }
      asm volatile("s_wait_asynccnt 0x0" ::: "memory");
    }
    __syncthreads();
    build_slab(ldsO, slab, n0, tid, 8);
    __syncthreads();

    const unsigned short* aptr = aRowBase + ph * KQ_;
    #pragma unroll 1
    for (int ks = 0; ks < KQ_ / 32; ++ks) {        // 18 k-steps per phase
      const int kb = ks * 32;
      v16bf A  = cat16(*(const v8bf*)(aptr + kb), *(const v8bf*)(aptr + kb + 8));
      v16bf Bm = cat16(*(const v8bf*)(bptr + kb), *(const v8bf*)(bptr + kb + 8));
      acc = __builtin_amdgcn_wmma_f32_16x16x32_bf16(false, A, false, Bm,
                                                    (short)0, acc, false, false);
    }
  }
  #pragma unroll
  for (int r = 0; r < 8; ++r) {
    int m = wv * 16 + r + lhi * 8;
    out[((size_t)img * 64 + m) * HW_ + n] = acc[r];
  }
}

// ---------------------------------------------------------------------------
// Launcher. ws layout (bytes):
//   wcat 1536*576*2 = 1,769,472
//   wub    64*4608*2 =   589,824
//   q,k  each 32*64*16384*2 = 67,108,864
//   vT   67,108,864
//   P    32*64*64*2 = 262,144
//   O    256*512*256*2 = 67,108,864          total ~271 MB
// ---------------------------------------------------------------------------
extern "C" void kernel_launch(void* const* d_in, const int* in_sizes, int n_in,
                              void* d_out, int out_size, void* d_ws, size_t ws_size,
                              hipStream_t stream) {
  (void)in_sizes; (void)n_in; (void)out_size; (void)ws_size;
  const float* x  = (const float*)d_in[0];
  const float* Wq = (const float*)d_in[1];
  const float* Wk = (const float*)d_in[2];
  const float* Wv = (const float*)d_in[3];
  const float* Wu = (const float*)d_in[4];

  char* ws = (char*)d_ws;
  unsigned short* wcat = (unsigned short*)(ws);
  unsigned short* wub  = (unsigned short*)(ws + 1769472);
  unsigned short* q    = (unsigned short*)(ws + 2359296);
  unsigned short* k    = (unsigned short*)(ws + 69468160);
  unsigned short* vt   = (unsigned short*)(ws + 136577024);
  unsigned short* p    = (unsigned short*)(ws + 203685888);
  unsigned short* O    = (unsigned short*)(ws + 203948032);
  float* out = (float*)d_out;

  pack_wcat<<<dim3((1536 * KQ_ + 255) / 256), 256, 0, stream>>>(Wq, Wk, Wv, wcat);
  pack_wu  <<<dim3((64 * KU_ + 255) / 256),   256, 0, stream>>>(Wu, wub);
  conv_qkv <<<dim3(256, 16), 256, 0, stream>>>(x, wcat, q, k, vt);
  scores_softmax<<<dim3(32), 512, 0, stream>>>(q, k, p);
  attn_out <<<dim3(512, 32), 256, 0, stream>>>(p, vt, O);
  conv_out <<<dim3(256, 16), 128, 0, stream>>>(O, wub, out);
}
